// GatedDeltaNet_8100308320394
// MI455X (gfx1250) — matrix operand order repaired and checked by reading, not compiled, not run on hardware
//
#include <hip/hip_runtime.h>
#include <cstdint>
#include <cstddef>

// ---------------- problem constants ----------------
#define B_       2
#define T_       2048
#define C_       2048
#define HK_      16
#define HV_      32
#define DK_      128
#define DV_      128
#define KW_      4
#define KEY_DIM_ 2048
#define VAL_DIM_ 4096
#define QKV_DIM_ 8192
#define CHUNK_   64
#define NCHUNK_  (T_ / CHUNK_)
#define MROWS_   (B_ * T_)   // 4096

typedef __attribute__((ext_vector_type(16))) __bf16 v16bf;
typedef __attribute__((ext_vector_type(8)))  __bf16 v8bf;
typedef __attribute__((ext_vector_type(8)))  float  v8f;
typedef __attribute__((ext_vector_type(4)))  unsigned int u32x4;
typedef __attribute__((ext_vector_type(8)))  int i32x8;
typedef __attribute__((ext_vector_type(4)))  int i32x4;

union FragBF {
  v16bf  v;
  v8bf   h[2];
  __bf16 e[16];
};

#if defined(__has_builtin)
#if __has_builtin(__builtin_amdgcn_tensor_load_to_lds) && __has_builtin(__builtin_amdgcn_s_wait_tensorcnt)
#define USE_TDM 1
#endif
#endif
#ifndef USE_TDM
#define USE_TDM 0
#endif

// ---- WMMA fragment loaders (CDNA5 16-bit layouts, wave32) ----
__device__ __forceinline__ v16bf ldA(const __bf16* base, int stride, int row0, int k0, int lane) {
  int m = lane & 15, hi = lane >> 4;
  const __bf16* p = base + (size_t)(row0 + m) * stride + k0 + hi * 16;
  FragBF f;
  f.h[0] = *(const v8bf*)(p);
  f.h[1] = *(const v8bf*)(p + 8);
  return f.v;
}
__device__ __forceinline__ v16bf ldB(const __bf16* base, int stride, int col0, int k0, int lane) {
  int n = lane & 15, hi = lane >> 4;
  const __bf16* p = base + (size_t)(col0 + n) * stride + k0 + hi * 16;
  FragBF f;
  f.h[0] = *(const v8bf*)(p);
  f.h[1] = *(const v8bf*)(p + 8);
  return f.v;
}
__device__ __forceinline__ v16bf ldB_f32(const float* base, int stride, int col0, int k0, int lane) {
  int n = lane & 15, hi = lane >> 4;
  const float* p = base + (size_t)(col0 + n) * stride + k0 + hi * 16;
  FragBF f;
#pragma unroll
  for (int i = 0; i < 16; ++i) f.e[i] = (__bf16)p[i];
  return f.v;
}

__device__ __forceinline__ v8f wmma_bf16(v16bf a, v16bf b, v8f c) {
  return __builtin_amdgcn_wmma_f32_16x16x32_bf16(false, a, false, b, (short)0, c, false, false);
}

__device__ __forceinline__ float sigmoidf_(float x) { return 1.f / (1.f + __expf(-x)); }
__device__ __forceinline__ float siluf_(float x)    { return x * sigmoidf_(x); }

#if USE_TDM
// ---- TDM: DMA a 64-row x 128-elem bf16 tile (row stride QKV_DIM) into LDS,
//      with LDS padding of 4 DWORDs every 64 DWORDs -> row stride 136 bf16. ----
__device__ __forceinline__ void tdm_load_tile_64x128(const __bf16* gsrc, void* lds_dst) {
  unsigned long long ga = (unsigned long long)(uintptr_t)gsrc;
  u32x4 g0;
  g0[0] = 1u;                                           // count=1, user mode
  g0[1] = (unsigned)(uintptr_t)lds_dst;                 // LDS byte address (addr[31:0])
  g0[2] = (unsigned)(ga & 0xFFFFFFFFu);                 // global_addr[31:0]
  g0[3] = (unsigned)((ga >> 32) & 0x01FFFFFFu) | (2u << 30);  // addr[56:32] | type=2
  i32x8 g1;
  g1[0] = (int)((1u << 16)        // data_size = 2 bytes
              | (1u << 20)        // pad_enable
              | (5u << 22)        // pad_interval: 64 DWORDs (256B)
              | (3u << 25));      // pad_amount: 4 DWORDs (16B)
  g1[1] = (int)(128u << 16);      // tensor_dim0 = 128 (lo16 in [31:16])
  g1[2] = (int)(64u << 16);       // tensor_dim0 hi=0 | tensor_dim1 = 64
  g1[3] = (int)(128u << 16);      // tensor_dim1 hi=0 | tile_dim0 = 128
  g1[4] = 64;                     // tile_dim1 = 64, tile_dim2 = 0
  g1[5] = QKV_DIM_;               // tensor_dim0_stride lo32
  g1[6] = 0;                      // stride0 hi | stride1 lo
  g1[7] = 0;
  i32x4 z4 = {0, 0, 0, 0};
#if defined(__clang_major__) && (__clang_major__ >= 23)
  i32x8 z8 = {0, 0, 0, 0, 0, 0, 0, 0};
  __builtin_amdgcn_tensor_load_to_lds(g0, g1, z4, z4, z8, 0);
#else
  __builtin_amdgcn_tensor_load_to_lds(g0, g1, z4, z4, 0);
#endif
}
#endif

// ---------------- fp32 -> bf16 convert ----------------
__global__ void cvt_f32_bf16_kernel(const float* __restrict__ src, __bf16* __restrict__ dst, long n) {
  long i = (long)blockIdx.x * blockDim.x + threadIdx.x;
  if (i < n) dst[i] = (__bf16)src[i];
}

// ---------------- big NT GEMM: one wave computes a 32x64 strip ----------------
// C[m,n] = sum_k A[m,k]*W[n,k]; grid = (N/64, M/32); block = 32 (one wave).
__global__ void __launch_bounds__(32)
gemm_nt_wmma_big(const __bf16* __restrict__ A, const __bf16* __restrict__ W,
                 float* __restrict__ Cf, __bf16* __restrict__ Cb,
                 int N, int K) {
  int lane = threadIdx.x & 31;
  int m  = lane & 15, hi = lane >> 4;
  size_t tm0 = (size_t)blockIdx.y * 32;
  size_t tn0 = (size_t)blockIdx.x * 64;
  const __bf16* ar0 = A + (tm0 + m) * (size_t)K;
  const __bf16* ar1 = ar0 + (size_t)16 * K;
  const __bf16* br0 = W + (tn0 + m) * (size_t)K;
  v8f c[2][4] = {};
  for (int k0 = 0; k0 < K; k0 += 32) {
    __builtin_prefetch(ar0 + k0 + 512, 0, 1);
    __builtin_prefetch(ar1 + k0 + 512, 0, 1);
    FragBF a0, a1;
    a0.h[0] = *(const v8bf*)(ar0 + k0 + hi * 16);
    a0.h[1] = *(const v8bf*)(ar0 + k0 + hi * 16 + 8);
    a1.h[0] = *(const v8bf*)(ar1 + k0 + hi * 16);
    a1.h[1] = *(const v8bf*)(ar1 + k0 + hi * 16 + 8);
#pragma unroll
    for (int j = 0; j < 4; ++j) {
      const __bf16* br = br0 + (size_t)(j * 16) * K;
      __builtin_prefetch(br + k0 + 512, 0, 1);
      FragBF b;
      b.h[0] = *(const v8bf*)(br + k0 + hi * 16);
      b.h[1] = *(const v8bf*)(br + k0 + hi * 16 + 8);
      c[0][j] = wmma_bf16(a0.v, b.v, c[0][j]);
      c[1][j] = wmma_bf16(a1.v, b.v, c[1][j]);
    }
  }
#pragma unroll
  for (int mi = 0; mi < 2; ++mi) {
#pragma unroll
    for (int j = 0; j < 4; ++j) {
      size_t row0 = tm0 + mi * 16 + hi * 8;
      size_t col  = tn0 + j * 16 + m;
#pragma unroll
      for (int e = 0; e < 8; ++e) {
        size_t o = (row0 + e) * N + col;
        if (Cf) Cf[o] = c[mi][j][e];
        if (Cb) Cb[o] = (__bf16)c[mi][j][e];
      }
    }
  }
}

// ---------------- small-N NT GEMM (wave per 16x16 tile) for b/a projections ----------------
__global__ void __launch_bounds__(32)
gemm_nt_wmma_small(const __bf16* __restrict__ A, const __bf16* __restrict__ W,
                   float* __restrict__ Cf, int N, int K) {
  int lane = threadIdx.x & 31;
  int tn = blockIdx.x, tm = blockIdx.y;
  int m  = lane & 15, hi = lane >> 4;
  const __bf16* ar = A + (size_t)(tm * 16 + m) * K;
  const __bf16* br = W + (size_t)(tn * 16 + m) * K;
  v8f c = {};
  for (int k0 = 0; k0 < K; k0 += 32) {
    FragBF a, b;
    a.h[0] = *(const v8bf*)(ar + k0 + hi * 16);
    a.h[1] = *(const v8bf*)(ar + k0 + hi * 16 + 8);
    b.h[0] = *(const v8bf*)(br + k0 + hi * 16);
    b.h[1] = *(const v8bf*)(br + k0 + hi * 16 + 8);
    c = wmma_bf16(a.v, b.v, c);
  }
  int col  = tn * 16 + m;
  int row0 = tm * 16 + hi * 8;
#pragma unroll
  for (int e = 0; e < 8; ++e) Cf[(size_t)(row0 + e) * N + col] = c[e];
}

// ---------------- depthwise causal conv1d (KW=4) + SiLU ----------------
__global__ void conv_silu_kernel(const __bf16* __restrict__ raw, const float* __restrict__ w,
                                 __bf16* __restrict__ out) {
  size_t idx = (size_t)blockIdx.x * blockDim.x + threadIdx.x;
  const size_t total = (size_t)B_ * T_ * QKV_DIM_;
  if (idx >= total) return;
  int ch = (int)(idx % QKV_DIM_);
  size_t bt = idx / QKV_DIM_;
  int t = (int)(bt % T_);
  float acc = 0.f;
#pragma unroll
  for (int j = 0; j < KW_; ++j) {
    int ts = t + j - (KW_ - 1);
    if (ts >= 0)
      acc += w[ch * KW_ + j] * (float)raw[(bt + j - (KW_ - 1)) * QKV_DIM_ + ch];
  }
  out[idx] = (__bf16)siluf_(acc);
}

// ---------------- q/k per-head L2 normalization (one wave per (b,t,head)) ----------------
__global__ void __launch_bounds__(32)
qk_norm_kernel(__bf16* __restrict__ qkv) {
  int id = blockIdx.x;                 // B*T*(2*HK)
  int head = id % (2 * HK_);
  size_t bt = (size_t)(id / (2 * HK_));
  int base = (head < HK_) ? head * DK_ : KEY_DIM_ + (head - HK_) * DK_;
  __bf16* p = qkv + bt * QKV_DIM_ + base;
  int lane = threadIdx.x & 31;
  float v[4];
  float ss = 0.f;
#pragma unroll
  for (int i = 0; i < 4; ++i) { v[i] = (float)p[lane + 32 * i]; ss += v[i] * v[i]; }
#pragma unroll
  for (int off = 16; off > 0; off >>= 1) ss += __shfl_xor(ss, off, 32);
  float sc = 1.f / fmaxf(sqrtf(ss), 1e-12f);
#pragma unroll
  for (int i = 0; i < 4; ++i) p[lane + 32 * i] = (__bf16)(v[i] * sc);
}

// ---------------- beta = sigmoid(b); g = -exp(A_log)*softplus(a + dt_bias) ----------------
__global__ void gate_kernel(const float* __restrict__ braw, const float* __restrict__ araw,
                            const float* __restrict__ dt_bias, const float* __restrict__ A_log,
                            float* __restrict__ beta, float* __restrict__ g) {
  size_t idx = (size_t)blockIdx.x * blockDim.x + threadIdx.x;
  const size_t total = (size_t)B_ * T_ * HV_;
  if (idx >= total) return;
  int h = (int)(idx % HV_);
  beta[idx] = sigmoidf_(braw[idx]);
  float av = araw[idx] + dt_bias[h];
  float sp = (av > 20.f) ? av : log1pf(__expf(av));
  g[idx] = -__expf(A_log[h]) * sp;
}

// ---------------- chunked gated linear-attention scan (WMMA + TDM staging) ----------------
#define SQS_ 136   // bf16 row stride for [64 x 128] chunk matrices (TDM pad produces this)
#define STS_ 72    // bf16 row stride for [128 x 64] transposed matrices / A
#define SSS_ 132   // f32  row stride for state [128 x 128]
__global__ void __launch_bounds__(128)
delta_scan_kernel(const __bf16* __restrict__ qkv, const float* __restrict__ g_arr,
                  const float* __restrict__ beta_arr, float* __restrict__ o_core) {
  extern __shared__ char smem_raw[];
  __bf16* sh_q  = (__bf16*)smem_raw;        // [64][SQS_]  q rows
  __bf16* sh_k  = sh_q  + 64 * SQS_;        // [64][SQS_]  k rows
  __bf16* sh_v  = sh_k  + 64 * SQS_;        // [64][SQS_]  v rows (staging)
  __bf16* sh_kT = sh_v  + 64 * SQS_;        // [128][STS_] k^T pre-scaled by wK[j]
  __bf16* sh_vT = sh_kT + 128 * STS_;       // [128][STS_] v^T
  __bf16* sh_A  = sh_vT + 128 * STS_;       // [64][STS_]  masked attention P (bf16)
  float*  sh_ST = (float*)(sh_A + 64 * STS_);   // [128][SSS_] state S^T (f32)
  float*  sh_Gs   = sh_ST + 128 * SSS_;
  float*  sh_beta = sh_Gs + 64;
  float*  sh_lamQ = sh_beta + 64;
  float*  sh_wK   = sh_lamQ + 64;
  float*  sh_lam  = sh_wK + 64;

  int tid  = threadIdx.x;
  int lane = tid & 31;
  int w    = tid >> 5;

  int bh = blockIdx.x;
  int b  = bh / HV_;
  int hv = bh % HV_;
  int hk = hv >> 1;                          // REPEAT = 2

  const __bf16* qg = qkv + (size_t)b * T_ * QKV_DIM_ + hk * DK_;
  const __bf16* kg = qkv + (size_t)b * T_ * QKV_DIM_ + KEY_DIM_ + hk * DK_;
  const __bf16* vg = qkv + (size_t)b * T_ * QKV_DIM_ + 2 * KEY_DIM_ + (size_t)hv * DV_;
  const float*  gb = g_arr    + (size_t)b * T_ * HV_ + hv;
  const float*  bb = beta_arr + (size_t)b * T_ * HV_ + hv;

  for (int i = tid; i < 128 * SSS_; i += 128) sh_ST[i] = 0.f;
  __syncthreads();

  for (int c = 0; c < NCHUNK_; ++c) {
    int t0 = c * CHUNK_;

    // ---- chunk scalars ----
    if (tid < 64) {
      sh_Gs[tid]   = gb[(size_t)(t0 + tid) * HV_];
      sh_beta[tid] = bb[(size_t)(t0 + tid) * HV_];
    }
    __syncthreads();
    if (tid == 0) {
      float run = 0.f;
      for (int i = 0; i < 64; ++i) { run += sh_Gs[i]; sh_Gs[i] = run; }
      sh_lam[0] = __expf(run);
    }
    __syncthreads();
    if (tid < 64) {
      float Gend = sh_Gs[63];
      sh_lamQ[tid] = __expf(sh_Gs[tid]);
      sh_wK[tid]   = __expf(Gend - sh_Gs[tid]) * sh_beta[tid];
    }
    __syncthreads();

    // ---- stage chunk q/k/v into LDS ----
#if USE_TDM
    if (tid == 0) {
      tdm_load_tile_64x128(qg + (size_t)t0 * QKV_DIM_, sh_q);
      tdm_load_tile_64x128(kg + (size_t)t0 * QKV_DIM_, sh_k);
      tdm_load_tile_64x128(vg + (size_t)t0 * QKV_DIM_, sh_v);
    }
    __builtin_amdgcn_s_wait_tensorcnt(0);
    __syncthreads();
#else
    for (int i = tid; i < 64 * 128; i += 128) {
      int r = i >> 7, d = i & 127;
      sh_q[r * SQS_ + d] = qg[(size_t)(t0 + r) * QKV_DIM_ + d];
      sh_k[r * SQS_ + d] = kg[(size_t)(t0 + r) * QKV_DIM_ + d];
      sh_v[r * SQS_ + d] = vg[(size_t)(t0 + r) * QKV_DIM_ + d];
    }
    __syncthreads();
#endif
    // ---- LDS->LDS transpose; fold decay*beta into k^T ----
    for (int i = tid; i < 64 * 128; i += 128) {
      int r = i >> 7, d = i & 127;
      sh_kT[d * STS_ + r] = (__bf16)((float)sh_k[r * SQS_ + d] * sh_wK[r]);
      sh_vT[d * STS_ + r] = sh_v[r * SQS_ + d];
    }
    __syncthreads();

    // ---- inter-chunk: acc = lamQ[i] * (Q @ S_prev) ----
    v8f acc[8];
    int mt = w;
    int hi = lane >> 4;
#pragma unroll
    for (int nt = 0; nt < 8; ++nt) {
      v8f cc = {};
      for (int k0 = 0; k0 < 128; k0 += 32) {
        v16bf a  = ldA(sh_q, SQS_, mt * 16, k0, lane);
        v16bf bm = ldB_f32(sh_ST, SSS_, nt * 16, k0, lane);
        cc = wmma_bf16(a, bm, cc);
      }
#pragma unroll
      for (int e = 0; e < 8; ++e) cc[e] *= sh_lamQ[mt * 16 + hi * 8 + e];
      acc[nt] = cc;
    }

    // ---- intra-chunk attention: P = mask(Q K^T) ----
#pragma unroll
    for (int nt = 0; nt < 4; ++nt) {
      v8f cc = {};
      for (int k0 = 0; k0 < 128; k0 += 32) {
        v16bf a  = ldA(sh_q, SQS_, mt * 16, k0, lane);
        v16bf bm = ldB(sh_k, SQS_, nt * 16, k0, lane);
        cc = wmma_bf16(a, bm, cc);
      }
      int col = nt * 16 + (lane & 15);
#pragma unroll
      for (int e = 0; e < 8; ++e) {
        int row = mt * 16 + hi * 8 + e;
        float p = (row >= col)
                      ? cc[e] * __expf(sh_Gs[row] - sh_Gs[col]) * sh_beta[col]
                      : 0.f;
        sh_A[row * STS_ + col] = (__bf16)p;
      }
    }
    __syncthreads();

    // ---- acc += P @ V ; write out ----
#pragma unroll
    for (int nt = 0; nt < 8; ++nt) {
      v8f cc = acc[nt];
      for (int k0 = 0; k0 < 64; k0 += 32) {
        v16bf a  = ldA(sh_A, STS_, mt * 16, k0, lane);
        v16bf bm = ldB(sh_vT, STS_, nt * 16, k0, lane);
        cc = wmma_bf16(a, bm, cc);
      }
      int col = nt * 16 + (lane & 15);
#pragma unroll
      for (int e = 0; e < 8; ++e) {
        int t = t0 + mt * 16 + hi * 8 + e;
        o_core[((size_t)(b * T_ + t) * HV_ + hv) * DV_ + col] = cc[e];
      }
    }
    __syncthreads();

    // ---- state update: S^T = lam * S^T + V^T @ K~ ----
    float lam = sh_lam[0];
    for (int tix = 0; tix < 16; ++tix) {
      int mt2 = w * 2 + (tix >> 3);
      int nt2 = tix & 7;
      v8f cc = {};
      for (int k0 = 0; k0 < 64; k0 += 32) {
        v16bf a  = ldA(sh_vT, STS_, mt2 * 16, k0, lane);
        v16bf bm = ldB(sh_kT, STS_, nt2 * 16, k0, lane);
        cc = wmma_bf16(a, bm, cc);
      }
      int col = nt2 * 16 + (lane & 15);
#pragma unroll
      for (int e = 0; e < 8; ++e) {
        int row = mt2 * 16 + hi * 8 + e;
        int o = row * SSS_ + col;
        sh_ST[o] = lam * sh_ST[o] + cc[e];
      }
    }
    __syncthreads();
  }
}

// ---------------- gated RMSNorm: o = norm_w * rms(o) * silu(z), bf16 out ----------------
__global__ void __launch_bounds__(32)
rms_gate_kernel(const float* __restrict__ o_core, const __bf16* __restrict__ z,
                const float* __restrict__ norm_w, __bf16* __restrict__ og) {
  int id = blockIdx.x;                  // B*T*HV
  int h = id % HV_;
  size_t bt = (size_t)(id / HV_);
  const float* p = o_core + (size_t)id * DV_;
  int lane = threadIdx.x & 31;
  float v[4];
  float ss = 0.f;
#pragma unroll
  for (int i = 0; i < 4; ++i) { v[i] = p[lane + 32 * i]; ss += v[i] * v[i]; }
#pragma unroll
  for (int off = 16; off > 0; off >>= 1) ss += __shfl_xor(ss, off, 32);
  float r = rsqrtf(ss / (float)DV_ + 1e-6f);
  const __bf16* zp = z + bt * VAL_DIM_ + (size_t)h * DV_;
  __bf16* op = og + bt * VAL_DIM_ + (size_t)h * DV_;
#pragma unroll
  for (int i = 0; i < 4; ++i) {
    int d = lane + 32 * i;
    float zz = (float)zp[d];
    op[d] = (__bf16)(norm_w[d] * v[i] * r * siluf_(zz));
  }
}

// ---------------- launcher ----------------
extern "C" void kernel_launch(void* const* d_in, const int* in_sizes, int n_in,
                              void* d_out, int out_size, void* d_ws, size_t ws_size,
                              hipStream_t stream) {
  (void)in_sizes; (void)n_in; (void)out_size; (void)ws_size;
  const float* x       = (const float*)d_in[0];
  // d_in[1] = positions (unused)
  const float* W_qkv   = (const float*)d_in[2];
  const float* W_z     = (const float*)d_in[3];
  const float* W_b     = (const float*)d_in[4];
  const float* W_a     = (const float*)d_in[5];
  const float* conv_w  = (const float*)d_in[6];
  const float* dt_bias = (const float*)d_in[7];
  const float* A_log   = (const float*)d_in[8];
  const float* norm_w  = (const float*)d_in[9];
  const float* W_out   = (const float*)d_in[10];
  float* out = (float*)d_out;

  char* ws = (char*)d_ws;
  size_t off = 0;
  auto alloc = [&](size_t bytes) -> char* {
    char* p = ws + off;
    off += (bytes + 255) & ~(size_t)255;
    return p;
  };

  __bf16* xb      = (__bf16*)alloc((size_t)MROWS_ * C_ * 2);
  __bf16* wqkv    = (__bf16*)alloc((size_t)QKV_DIM_ * C_ * 2);
  __bf16* wz      = (__bf16*)alloc((size_t)VAL_DIM_ * C_ * 2);
  __bf16* wb      = (__bf16*)alloc((size_t)HV_ * C_ * 2);
  __bf16* wa      = (__bf16*)alloc((size_t)HV_ * C_ * 2);
  __bf16* wout    = (__bf16*)alloc((size_t)C_ * VAL_DIM_ * 2);
  __bf16* qkv_raw = (__bf16*)alloc((size_t)MROWS_ * QKV_DIM_ * 2);
  __bf16* qkv_pst = (__bf16*)alloc((size_t)MROWS_ * QKV_DIM_ * 2);
  __bf16* z_raw   = (__bf16*)alloc((size_t)MROWS_ * VAL_DIM_ * 2);
  float*  b_f32   = (float*)alloc((size_t)MROWS_ * HV_ * 4);
  float*  a_f32   = (float*)alloc((size_t)MROWS_ * HV_ * 4);
  float*  beta    = (float*)alloc((size_t)MROWS_ * HV_ * 4);
  float*  g       = (float*)alloc((size_t)MROWS_ * HV_ * 4);
  float*  o_core  = (float*)alloc((size_t)MROWS_ * VAL_DIM_ * 4);
  __bf16* o_gated = (__bf16*)alloc((size_t)MROWS_ * VAL_DIM_ * 2);

  auto cvt = [&](const float* s, __bf16* d, long n) {
    cvt_f32_bf16_kernel<<<dim3((unsigned)((n + 255) / 256)), dim3(256), 0, stream>>>(s, d, n);
  };
  cvt(x,     xb,   (long)MROWS_ * C_);
  cvt(W_qkv, wqkv, (long)QKV_DIM_ * C_);
  cvt(W_z,   wz,   (long)VAL_DIM_ * C_);
  cvt(W_b,   wb,   (long)HV_ * C_);
  cvt(W_a,   wa,   (long)HV_ * C_);
  cvt(W_out, wout, (long)C_ * VAL_DIM_);

  // big projections: 32x64 per wave
  gemm_nt_wmma_big<<<dim3(QKV_DIM_ / 64, MROWS_ / 32), dim3(32), 0, stream>>>(
      xb, wqkv, nullptr, qkv_raw, QKV_DIM_, C_);
  gemm_nt_wmma_big<<<dim3(VAL_DIM_ / 64, MROWS_ / 32), dim3(32), 0, stream>>>(
      xb, wz, nullptr, z_raw, VAL_DIM_, C_);
  // small b/a projections
  gemm_nt_wmma_small<<<dim3(HV_ / 16, MROWS_ / 16), dim3(32), 0, stream>>>(
      xb, wb, b_f32, HV_, C_);
  gemm_nt_wmma_small<<<dim3(HV_ / 16, MROWS_ / 16), dim3(32), 0, stream>>>(
      xb, wa, a_f32, HV_, C_);

  // depthwise causal conv + SiLU
  {
    size_t total = (size_t)B_ * T_ * QKV_DIM_;
    conv_silu_kernel<<<dim3((unsigned)((total + 255) / 256)), dim3(256), 0, stream>>>(
        qkv_raw, conv_w, qkv_pst);
  }
  // q/k L2 normalization
  qk_norm_kernel<<<dim3((unsigned)(B_ * T_ * 2 * HK_)), dim3(32), 0, stream>>>(qkv_pst);

  // gates
  {
    size_t total = (size_t)B_ * T_ * HV_;
    gate_kernel<<<dim3((unsigned)((total + 255) / 256)), dim3(256), 0, stream>>>(
        b_f32, a_f32, dt_bias, A_log, beta, g);
  }

  // chunked gated linear-attention scan
  {
    constexpr size_t sh_bf = ((size_t)3 * 64 * SQS_ + (size_t)2 * 128 * STS_ + (size_t)64 * STS_) * 2;
    constexpr size_t sh_sc = (size_t)128 * SSS_ * 4 + (size_t)(4 * 64 + 4) * 4;
    constexpr size_t shmem = sh_bf + sh_sc;  // ~163 KB (WGP LDS is 320 KB)
    delta_scan_kernel<<<dim3(B_ * HV_), dim3(128), shmem, stream>>>(qkv_pst, g, beta, o_core);
  }

  // gated RMSNorm
  rms_gate_kernel<<<dim3((unsigned)(B_ * T_ * HV_)), dim3(32), 0, stream>>>(
      o_core, z_raw, norm_w, o_gated);

  // output projection -> f32 d_out
  gemm_nt_wmma_big<<<dim3(C_ / 64, MROWS_ / 32), dim3(32), 0, stream>>>(
      o_gated, wout, out, nullptr, C_, VAL_DIM_);
}
